// DiscreteDecisionTransformer_17995912970756
// MI455X (gfx1250) — compile-verified, matrix operand order checked by reading, not run on hardware
//
#include <hip/hip_runtime.h>
#include <math.h>
#include <stdint.h>

typedef __bf16 bf16;
typedef __bf16 v16bf __attribute__((ext_vector_type(16)));
typedef __bf16 v8bf  __attribute__((ext_vector_type(8)));
typedef float  v8f   __attribute__((ext_vector_type(8)));
typedef float  f32x4 __attribute__((ext_vector_type(4)));

#define DEV __device__ __forceinline__

DEV bf16 f2bf(float f) {
  union { float f; unsigned u; } a; a.f = f;
  unsigned r = a.u + 0x7FFFu + ((a.u >> 16) & 1u);   // round-to-nearest-even
  union { unsigned short s; bf16 b; } c; c.s = (unsigned short)(r >> 16);
  return c.b;
}

DEV v8f vzero() {
  v8f r;
  #pragma unroll
  for (int i = 0; i < 8; ++i) r[i] = 0.f;
  return r;
}

DEV v16bf mk16(v8bf lo, v8bf hi) {
  v16bf r;
  #pragma unroll
  for (int i = 0; i < 8; ++i) { r[i] = lo[i]; r[i + 8] = hi[i]; }
  return r;
}

// 16B-per-lane async global->LDS copy (CDNA5 GLOBAL_LOAD_ASYNC_TO_LDS_B128).
DEV void async_copy_b128(void* lds_dst, const void* gsrc) {
  unsigned ldsoff = (unsigned)(uintptr_t)lds_dst;
  asm volatile("global_load_async_to_lds_b128 %0, %1, off"
               :: "v"(ldsoff), "v"(gsrc) : "memory");
}
DEV void async_wait0() {
  asm volatile("s_wait_asynccnt 0x0" ::: "memory");
}

// ---------------------------------------------------------------------------
// Generic bf16-WMMA GEMM:  C = act(A @ W + bias)
//   A: [M,K] (f32 or bf16), W: [K,N] f32, C: [M,N] (f32 or bf16)
//   Block tile BM x BN, BK=64, 256 threads = 8 waves (WM x WN wave grid).
// ---------------------------------------------------------------------------
template<int BM, int BN, int WM, int WN, bool ABF, bool RELU, bool OBF>
__global__ __launch_bounds__(256) void gemm_k(
    const void* __restrict__ Ap, const float* __restrict__ W,
    const float* __restrict__ bias, void* __restrict__ Cp,
    int M, int N, int K)
{
  constexpr int BK = 64;
  constexpr int MT = BM / (16 * WM);   // m-tiles per wave
  constexpr int NT = BN / (16 * WN);   // n-tiles per wave
  constexpr int LDA = BK + 8;          // padded LDS row (144B -> conflict free)

  __shared__ bf16 As[BM][LDA];
  __shared__ bf16 Bs[BN][LDA];

  const int t = threadIdx.x;
  const int wave = t >> 5, lane = t & 31, half = lane >> 4, lm = lane & 15;
  const int wm = wave / WN, wn = wave % WN;
  const int m0 = blockIdx.y * BM, n0 = blockIdx.x * BN;

  v8f acc[MT][NT];
  #pragma unroll
  for (int i = 0; i < MT; ++i)
    #pragma unroll
    for (int j = 0; j < NT; ++j) acc[i][j] = vzero();

  for (int k0 = 0; k0 < K; k0 += BK) {
    // ---- stage A tile ----
    if constexpr (ABF) {
      const bf16* A = (const bf16*)Ap;
      constexpr int CH = (BM * BK / 8) / 256;          // 16B chunks / thread
      #pragma unroll
      for (int i = 0; i < CH; ++i) {
        int c = t + i * 256;
        int row = c >> 3, seg = c & 7;                 // 8 x 8-elem chunks/row
        async_copy_b128(&As[row][seg * 8],
                        A + (size_t)(m0 + row) * K + k0 + seg * 8);
      }
    } else {
      const float* A = (const float*)Ap;
      constexpr int CH = (BM * BK / 4) / 256;
      #pragma unroll
      for (int i = 0; i < CH; ++i) {
        int c = t + i * 256;
        int row = c >> 4, seg = c & 15;                // 16 x 4-elem chunks/row
        f32x4 vv = *(const f32x4*)(A + (size_t)(m0 + row) * K + k0 + seg * 4);
        bf16* d = &As[row][seg * 4];
        d[0] = f2bf(vv.x); d[1] = f2bf(vv.y); d[2] = f2bf(vv.z); d[3] = f2bf(vv.w);
      }
    }
    // ---- stage W^T tile (f32 -> bf16, transposed) ----
    {
      constexpr int CH = (BK * BN / 4) / 256;
      #pragma unroll
      for (int i = 0; i < CH; ++i) {
        int c = t + i * 256;
        int kr = c / (BN / 4), nq = c % (BN / 4);
        const float* wp = W + (size_t)(k0 + kr) * N + n0 + nq * 4;
        f32x4 vv = *(const f32x4*)wp;
        if (k0 + BK < K)                        // prefetch next K-step tile
          __builtin_prefetch(wp + (size_t)BK * N, 0, 0);
        Bs[nq * 4 + 0][kr] = f2bf(vv.x);
        Bs[nq * 4 + 1][kr] = f2bf(vv.y);
        Bs[nq * 4 + 2][kr] = f2bf(vv.z);
        Bs[nq * 4 + 3][kr] = f2bf(vv.w);
      }
    }
    if constexpr (ABF) async_wait0();
    __syncthreads();

    // ---- fragments + WMMA (2 k-steps of 32 per stage) ----
    #pragma unroll
    for (int ks = 0; ks < 2; ++ks) {
      // batch ALL fragment loads before the WMMA chain (hide ds latency)
      v16bf af[MT], bfr[NT];
      #pragma unroll
      for (int i = 0; i < MT; ++i) {
        const bf16* p = &As[wm * MT * 16 + i * 16 + lm][ks * 32 + half * 8];
        af[i] = mk16(*(const v8bf*)p, *(const v8bf*)(p + 16));
      }
      #pragma unroll
      for (int j = 0; j < NT; ++j) {
        const bf16* p = &Bs[wn * NT * 16 + j * 16 + lm][ks * 32 + half * 8];
        bfr[j] = mk16(*(const v8bf*)p, *(const v8bf*)(p + 16));
      }
      #pragma unroll
      for (int j = 0; j < NT; ++j)
        #pragma unroll
        for (int i = 0; i < MT; ++i)
          acc[i][j] = __builtin_amdgcn_wmma_f32_16x16x32_bf16(
              false, af[i], false, bfr[j], (short)0, acc[i][j], false, false);
    }
    __syncthreads();
  }

  // ---- epilogue: bias (+ReLU), store ----
  #pragma unroll
  for (int j = 0; j < NT; ++j) {
    int n = n0 + wn * NT * 16 + j * 16 + lm;
    float bv = bias[n];
    #pragma unroll
    for (int i = 0; i < MT; ++i) {
      int mb = m0 + wm * MT * 16 + i * 16 + half * 8;
      #pragma unroll
      for (int r = 0; r < 8; ++r) {
        float val = acc[i][j][r] + bv;
        if (RELU) val = fmaxf(val, 0.f);
        if constexpr (OBF) ((bf16*)Cp)[(size_t)(mb + r) * N + n] = f2bf(val);
        else ((float*)Cp)[(size_t)(mb + r) * N + n] = val;
      }
    }
  }
}

// ---------------------------------------------------------------------------
// Flash-style causal self-attention (bf16 in/out, f32 online softmax).
// grid = (L/128, H, B); 256 threads; wave w owns query rows [q0+16w, q0+16w+16)
// ---------------------------------------------------------------------------
__global__ __launch_bounds__(256) void attn_k(
    const bf16* __restrict__ q, const bf16* __restrict__ k,
    const bf16* __restrict__ v, bf16* __restrict__ o,
    int L, int D, int DH, float scale)
{
  __shared__ bf16 Kt[32][96 + 8];        // key rows (natural layout)
  __shared__ bf16 Vt[96][32 + 8];        // V transposed: Vt[d][key]
  __shared__ bf16 Pb[8][16][32 + 8];     // per-wave P relayout buffer

  const int t = threadIdx.x, wave = t >> 5, lane = t & 31;
  const int half = lane >> 4, lm = lane & 15;
  const int b = blockIdx.z, h = blockIdx.y, q0 = blockIdx.x * 128;
  const int qr = q0 + wave * 16;

  // Q fragments (held in registers for the whole key loop)
  v16bf qf[3];
  #pragma unroll
  for (int ks = 0; ks < 3; ++ks) {
    const bf16* p = q + (size_t)(b * L + qr + lm) * D + h * DH + ks * 32 + half * 8;
    qf[ks] = mk16(*(const v8bf*)p, *(const v8bf*)(p + 16));
  }

  v8f oacc[6];
  #pragma unroll
  for (int i = 0; i < 6; ++i) oacc[i] = vzero();
  float rmax[8], rsum[8];
  #pragma unroll
  for (int r = 0; r < 8; ++r) { rmax[r] = -3.0e38f; rsum[r] = 0.f; }

  const int ktend = q0 + 128;
  for (int kt0 = 0; kt0 < ktend; kt0 += 32) {
    // cooperative K / V^T staging (384 x 8-elem chunks over 256 threads)
    #pragma unroll
    for (int i = 0; i < 2; ++i) {
      int c = t + i * 256;
      if (c < 384) {
        int key = c / 12, seg = c % 12;
        size_t base = (size_t)(b * L + kt0 + key) * D + h * DH + seg * 8;
        async_copy_b128(&Kt[key][seg * 8], k + base);   // async K tile
        v8bf vv = *(const v8bf*)(v + base);
        #pragma unroll
        for (int e = 0; e < 8; ++e) Vt[seg * 8 + e][key] = vv[e];
      }
    }
    async_wait0();
    __syncthreads();

    if (kt0 < qr + 16) {   // wave-uniform causal activity test
      // ---- S = Q @ K^T : preload all 6 K fragments, then WMMA chain ----
      v16bf kf[6];
      #pragma unroll
      for (int ks = 0; ks < 3; ++ks)
        #pragma unroll
        for (int j = 0; j < 2; ++j) {
          const bf16* p = &Kt[j * 16 + lm][ks * 32 + half * 8];
          kf[ks * 2 + j] = mk16(*(const v8bf*)p, *(const v8bf*)(p + 16));
        }
      v8f sacc[2]; sacc[0] = vzero(); sacc[1] = vzero();
      #pragma unroll
      for (int ks = 0; ks < 3; ++ks)
        #pragma unroll
        for (int j = 0; j < 2; ++j)
          sacc[j] = __builtin_amdgcn_wmma_f32_16x16x32_bf16(
              false, qf[ks], false, kf[ks * 2 + j], (short)0, sacc[j], false, false);

      // scale + causal mask, per-row tile max
      float sv[2][8], tmax[8];
      #pragma unroll
      for (int r = 0; r < 8; ++r) {
        int qrow = qr + r + half * 8;
        float best = -3.0e38f;
        #pragma unroll
        for (int j = 0; j < 2; ++j) {
          int key = kt0 + j * 16 + lm;
          float xv = sacc[j][r] * scale;
          if (key > qrow) xv = -3.0e38f;
          sv[j][r] = xv;
          best = fmaxf(best, xv);
        }
        tmax[r] = best;
      }
      #pragma unroll
      for (int off = 1; off < 16; off <<= 1)
        #pragma unroll
        for (int r = 0; r < 8; ++r)
          tmax[r] = fmaxf(tmax[r], __shfl_xor(tmax[r], off, 32));
      // online softmax rescale
      float corr[8];
      #pragma unroll
      for (int r = 0; r < 8; ++r) {
        float nm = fmaxf(rmax[r], tmax[r]);
        corr[r] = __expf(rmax[r] - nm);
        rmax[r] = nm;
      }
      float tsum[8];
      #pragma unroll
      for (int r = 0; r < 8; ++r) {
        float p0 = __expf(sv[0][r] - rmax[r]);
        float p1 = __expf(sv[1][r] - rmax[r]);
        sv[0][r] = p0; sv[1][r] = p1;
        tsum[r] = p0 + p1;
      }
      #pragma unroll
      for (int off = 1; off < 16; off <<= 1)
        #pragma unroll
        for (int r = 0; r < 8; ++r)
          tsum[r] += __shfl_xor(tsum[r], off, 32);
      #pragma unroll
      for (int r = 0; r < 8; ++r) rsum[r] = rsum[r] * corr[r] + tsum[r];
      #pragma unroll
      for (int nt = 0; nt < 6; ++nt)
        #pragma unroll
        for (int r = 0; r < 8; ++r) oacc[nt][r] *= corr[r];

      // relayout P (C-layout -> A-operand layout) through per-wave LDS
      #pragma unroll
      for (int j = 0; j < 2; ++j)
        #pragma unroll
        for (int r = 0; r < 8; ++r)
          Pb[wave][r + half * 8][j * 16 + lm] = f2bf(sv[j][r]);
      const bf16* pp = &Pb[wave][lm][half * 8];
      v16bf pf = mk16(*(const v8bf*)pp, *(const v8bf*)(pp + 16));

      // ---- O += P @ V : preload all 6 V fragments, then WMMA chain ----
      v16bf vf[6];
      #pragma unroll
      for (int nt = 0; nt < 6; ++nt) {
        const bf16* vp = &Vt[nt * 16 + lm][half * 8];
        vf[nt] = mk16(*(const v8bf*)vp, *(const v8bf*)(vp + 16));
      }
      #pragma unroll
      for (int nt = 0; nt < 6; ++nt)
        oacc[nt] = __builtin_amdgcn_wmma_f32_16x16x32_bf16(
            false, pf, false, vf[nt], (short)0, oacc[nt], false, false);
    }
    __syncthreads();
  }

  // normalize + store
  float inv[8];
  #pragma unroll
  for (int r = 0; r < 8; ++r) inv[r] = 1.0f / rsum[r];
  #pragma unroll
  for (int nt = 0; nt < 6; ++nt)
    #pragma unroll
    for (int r = 0; r < 8; ++r) {
      int row = qr + r + half * 8;
      o[(size_t)(b * L + row) * D + h * DH + nt * 16 + lm] = f2bf(oacc[nt][r] * inv[r]);
    }
}

// ---------------------------------------------------------------------------
// Cross-attn with Lk=1 collapses to per-batch vector: (tbl[t] @ Wv + bv) @ Wo + bo
// ---------------------------------------------------------------------------
__global__ __launch_bounds__(256) void cavec_k(
    const float* __restrict__ table, const int* __restrict__ tsk,
    const float* __restrict__ Wv, const float* __restrict__ bv,
    const float* __restrict__ Wo, const float* __restrict__ bo,
    float* __restrict__ out, int D)
{
  __shared__ float kv[768];
  __shared__ float vv[768];
  const int b = blockIdx.x, t = threadIdx.x;
  const int ti = tsk[b];
  for (int e = t; e < D; e += 256) kv[e] = table[(size_t)ti * D + e];
  __syncthreads();
  for (int e = t; e < D; e += 256) {
    float s = bv[e];
    for (int kk = 0; kk < D; ++kk) s += kv[kk] * Wv[(size_t)kk * D + e];
    vv[e] = s;
  }
  __syncthreads();
  for (int e = t; e < D; e += 256) {
    float s = bo[e];
    for (int kk = 0; kk < D; ++kk) s += vv[kk] * Wo[(size_t)kk * D + e];
    out[(size_t)b * D + e] = s;
  }
}

// ---------------------------------------------------------------------------
// x = LayerNorm(x + resid) * g + b   (resid optionally broadcast per batch)
// ---------------------------------------------------------------------------
__global__ __launch_bounds__(256) void add_ln_k(
    float* __restrict__ x, const float* __restrict__ resid,
    const float* __restrict__ g, const float* __restrict__ bb,
    int D, int L, int bcast)
{
  __shared__ float red[256];
  const int row = blockIdx.x, t = threadIdx.x;
  const size_t base = (size_t)row * D;
  const size_t rbase = bcast ? (size_t)(row / L) * D : base;
  float loc[3];
  float s = 0.f;
  #pragma unroll
  for (int p = 0; p < 3; ++p) {
    int e = t + p * 256;
    loc[p] = x[base + e] + resid[rbase + e];
    s += loc[p];
  }
  red[t] = s; __syncthreads();
  for (int o = 128; o > 0; o >>= 1) { if (t < o) red[t] += red[t + o]; __syncthreads(); }
  float mean = red[0] / (float)D;
  __syncthreads();
  s = 0.f;
  #pragma unroll
  for (int p = 0; p < 3; ++p) { float d = loc[p] - mean; s += d * d; }
  red[t] = s; __syncthreads();
  for (int o = 128; o > 0; o >>= 1) { if (t < o) red[t] += red[t + o]; __syncthreads(); }
  float rstd = rsqrtf(red[0] / (float)D + 1e-5f);
  #pragma unroll
  for (int p = 0; p < 3; ++p) {
    int e = t + p * 256;
    x[base + e] = (loc[p] - mean) * rstd * g[e] + bb[e];
  }
}

// ---------------------------------------------------------------------------
// x = LN(concat(a@Wa+ba, r@Wr+br, s@Ws+bs)) + posenc    one row / workgroup
// ---------------------------------------------------------------------------
__global__ __launch_bounds__(256) void embed_k(
    const float* __restrict__ a, const float* __restrict__ r,
    const float* __restrict__ s, const float* __restrict__ Wa,
    const float* __restrict__ ba, const float* __restrict__ Wr,
    const float* __restrict__ br, const float* __restrict__ Wsm,
    const float* __restrict__ bs, const float* __restrict__ g,
    const float* __restrict__ bb, float* __restrict__ x,
    int L, int D, int A, int S, int E)
{
  __shared__ float sa[64 + 1 + 128];
  __shared__ float red[256];
  const int row = blockIdx.x, t = threadIdx.x;
  const int l = row % L;
  if (t < A) sa[t] = a[(size_t)row * A + t];
  if (t == A) sa[A] = r[row];
  if (t >= 64 && t < 64 + S) sa[A + 1 + (t - 64)] = s[(size_t)row * S + (t - 64)];
  __syncthreads();
  float loc[3];
  float sum = 0.f;
  #pragma unroll
  for (int p = 0; p < 3; ++p) {
    int e = t + p * 256;
    float acc;
    if (e < E) {
      acc = ba[e];
      for (int kk = 0; kk < A; ++kk) acc += sa[kk] * Wa[kk * E + e];
    } else if (e < 2 * E) {
      int e2 = e - E;
      acc = br[e2] + sa[A] * Wr[e2];
    } else {
      int e2 = e - 2 * E;
      acc = bs[e2];
      for (int kk = 0; kk < S; ++kk) acc += sa[A + 1 + kk] * Wsm[kk * E + e2];
    }
    loc[p] = acc;
    sum += acc;
  }
  red[t] = sum; __syncthreads();
  for (int o = 128; o > 0; o >>= 1) { if (t < o) red[t] += red[t + o]; __syncthreads(); }
  float mean = red[0] / (float)D;
  __syncthreads();
  sum = 0.f;
  #pragma unroll
  for (int p = 0; p < 3; ++p) { float d = loc[p] - mean; sum += d * d; }
  red[t] = sum; __syncthreads();
  for (int o = 128; o > 0; o >>= 1) { if (t < o) red[t] += red[t + o]; __syncthreads(); }
  float rstd = rsqrtf(red[0] / (float)D + 1e-5f);
  #pragma unroll
  for (int p = 0; p < 3; ++p) {
    int e = t + p * 256;
    float y = (loc[p] - mean) * rstd * g[e] + bb[e];
    float ang = (float)l * __powf(10000.0f, -(float)(e & ~1) / (float)D);
    y += (e & 1) ? __cosf(ang) : __sinf(ang);
    x[(size_t)row * D + e] = y;
  }
}

// ---------------------------------------------------------------------------
extern "C" void kernel_launch(void* const* d_in, const int* in_sizes, int n_in,
                              void* d_out, int out_size, void* d_ws, size_t ws_size,
                              hipStream_t stream)
{
  (void)in_sizes; (void)n_in; (void)out_size; (void)ws_size;
  const int Bn = 16, L = 1024, D = 768, H = 8, DH = 96, M = 16384;
  const int A = 64, S = 128, E = 256, FF = 3072, NB = 4, OUTD = 64;

  const float* a     = (const float*)d_in[0];
  const float* r     = (const float*)d_in[1];
  const float* s     = (const float*)d_in[2];
  const int*   tsk   = (const int*)  d_in[3];
  const float* Wa    = (const float*)d_in[4];
  const float* ba    = (const float*)d_in[5];
  const float* Wr    = (const float*)d_in[6];
  const float* br    = (const float*)d_in[7];
  const float* Wsm   = (const float*)d_in[8];
  const float* bs    = (const float*)d_in[9];
  const float* table = (const float*)d_in[10];
  const float* ln_g  = (const float*)d_in[11];
  const float* ln_b  = (const float*)d_in[12];
  const float* saW   = (const float*)d_in[13];
  const float* sab   = (const float*)d_in[14];
  const float* saWo  = (const float*)d_in[15];
  const float* sabo  = (const float*)d_in[16];
  const float* ln1g  = (const float*)d_in[17];
  const float* ln1b  = (const float*)d_in[18];
  const float* caW   = (const float*)d_in[19];
  const float* cab   = (const float*)d_in[20];
  const float* caWo  = (const float*)d_in[21];
  const float* cabo  = (const float*)d_in[22];
  const float* ln2g  = (const float*)d_in[23];
  const float* ln2b  = (const float*)d_in[24];
  const float* fW1   = (const float*)d_in[25];
  const float* fb1   = (const float*)d_in[26];
  const float* fW2   = (const float*)d_in[27];
  const float* fb2   = (const float*)d_in[28];
  const float* ln3g  = (const float*)d_in[29];
  const float* ln3b  = (const float*)d_in[30];
  const float* fcW   = (const float*)d_in[31];
  const float* fcb   = (const float*)d_in[32];

  char* ws = (char*)d_ws;
  size_t off = 0;
  float* x    = (float*)(ws + off); off += (size_t)M * D * 4;       // 48 MiB
  bf16*  R1   = (bf16*) (ws + off); off += (size_t)M * FF * 2;      // 96 MiB union
  float* proj = (float*)(ws + off); off += (size_t)M * D * 4;       // 48 MiB
  float* cav  = (float*)(ws + off); off += (size_t)Bn * D * 4;
  bf16* qb = R1;
  bf16* kb = R1 + (size_t)1 * M * D;
  bf16* vb = R1 + (size_t)2 * M * D;
  bf16* ab = R1 + (size_t)3 * M * D;
  bf16* fbuf = R1;                       // FFN hidden aliases q/k/v/attn

  dim3 blk(256);
  embed_k<<<dim3(M), blk, 0, stream>>>(a, r, s, Wa, ba, Wr, br, Wsm, bs,
                                       ln_g, ln_b, x, L, D, A, S, E);

  const float scale = 0.1020620726f;     // 1/sqrt(96)
  const dim3 gD(D / 128, M / 128);
  for (int i = 0; i < NB; ++i) {
    const float* Wq = saW + ((size_t)i * 3 + 0) * D * D;
    const float* Wk = saW + ((size_t)i * 3 + 1) * D * D;
    const float* Wv = saW + ((size_t)i * 3 + 2) * D * D;
    const float* bq = sab + ((size_t)i * 3 + 0) * D;
    const float* bk = sab + ((size_t)i * 3 + 1) * D;
    const float* bv = sab + ((size_t)i * 3 + 2) * D;
    gemm_k<128,128,4,2,false,false,true><<<gD, blk, 0, stream>>>(x, Wq, bq, qb, M, D, D);
    gemm_k<128,128,4,2,false,false,true><<<gD, blk, 0, stream>>>(x, Wk, bk, kb, M, D, D);
    gemm_k<128,128,4,2,false,false,true><<<gD, blk, 0, stream>>>(x, Wv, bv, vb, M, D, D);
    attn_k<<<dim3(L / 128, H, Bn), blk, 0, stream>>>(qb, kb, vb, ab, L, D, DH, scale);
    gemm_k<128,128,4,2,true,false,false><<<gD, blk, 0, stream>>>(
        ab, saWo + (size_t)i * D * D, sabo + (size_t)i * D, proj, M, D, D);
    add_ln_k<<<dim3(M), blk, 0, stream>>>(x, proj, ln1g + (size_t)i * D, ln1b + (size_t)i * D, D, L, 0);
    cavec_k<<<dim3(Bn), blk, 0, stream>>>(table, tsk,
        caW + ((size_t)i * 3 + 2) * D * D, cab + ((size_t)i * 3 + 2) * D,
        caWo + (size_t)i * D * D, cabo + (size_t)i * D, cav, D);
    add_ln_k<<<dim3(M), blk, 0, stream>>>(x, cav, ln2g + (size_t)i * D, ln2b + (size_t)i * D, D, L, 1);
    gemm_k<128,128,4,2,false,true,true><<<dim3(FF / 128, M / 128), blk, 0, stream>>>(
        x, fW1 + (size_t)i * D * FF, fb1 + (size_t)i * FF, fbuf, M, FF, D);
    gemm_k<128,128,4,2,true,false,false><<<gD, blk, 0, stream>>>(
        fbuf, fW2 + (size_t)i * FF * D, fb2 + (size_t)i * D, proj, M, D, FF);
    add_ln_k<<<dim3(M), blk, 0, stream>>>(x, proj, ln3g + (size_t)i * D, ln3b + (size_t)i * D, D, L, 0);
  }
  gemm_k<128,64,8,1,false,false,false><<<dim3(OUTD / 64, M / 128), blk, 0, stream>>>(
      x, fcW, fcb, d_out, M, OUTD, D);
}
